// FEARec_23407571763913
// MI455X (gfx1250) — matrix operand order repaired and checked by reading, not compiled, not run on hardware
//
#include <hip/hip_runtime.h>
#include <hip/hip_bf16.h>
#include <math.h>

// ---------------------------------------------------------------------------
// FEARec forward on MI455X (gfx1250, wave32, WMMA).
// All GEMMs / attention matmuls use v_wmma_f32_16x16x32_bf16 (f32 accum).
// LDS tiles are stored so each lane's 16 fragment elements are two 8-element runs (k=8h+s, 16+8h+s)
// (fragment loads = 2x ds_load_b128), and LDS fills use float4 global loads
// + packed bf16x4 stores so loads batch under a single s_wait_loadcnt.
// ---------------------------------------------------------------------------

typedef __attribute__((ext_vector_type(16))) __bf16 bf16x16;
typedef __attribute__((ext_vector_type(8)))  __bf16 bf16x8;
typedef __attribute__((ext_vector_type(4)))  __bf16 bf16x4;
typedef __attribute__((ext_vector_type(8)))  float  f32x8;

#define Bq 4
#define Sq 1024
#define HID 256
#define NHEAD 4
#define DHEAD 64
#define TOPK 69
#define NFREQ 33

#define MODE_PLAIN 0
#define MODE_GELU  1
#define MODE_HEADS 2

#define SHUF16(a, b) __builtin_shufflevector((a), (b), 0,1,2,3,4,5,6,7,8,9,10,11,12,13,14,15)

__device__ __forceinline__ bf16x16 ldfrag(const __bf16* p0, const __bf16* p1) {
  bf16x8 a = *(const bf16x8*)p0;
  bf16x8 b = *(const bf16x8*)p1;
  return SHUF16(a, b);
}

__device__ __forceinline__ bf16x4 cvt4(float4 v) {
  bf16x4 r;
  r[0] = (__bf16)v.x; r[1] = (__bf16)v.y; r[2] = (__bf16)v.z; r[3] = (__bf16)v.w;
  return r;
}

// ---------------- block-wide sum over 256 threads (8 waves) ----------------
__device__ __forceinline__ float blockSum256(float v, float* red) {
  #pragma unroll
  for (int off = 16; off; off >>= 1) v += __shfl_xor(v, off, 32);
  const int wv = threadIdx.x >> 5, lane = threadIdx.x & 31;
  __syncthreads();
  if (lane == 0) red[wv] = v;
  __syncthreads();
  float r = 0.f;
  #pragma unroll
  for (int i = 0; i < 8; ++i) r += red[i];
  __syncthreads();
  return r;
}

// ---------------- embedding + pos + layernorm(eps=1e-12) -------------------
__global__ void __launch_bounds__(256)
embed_ln_kernel(const int* __restrict__ log_seqs, const float* __restrict__ item_emb,
                const float* __restrict__ pos_emb, const float* __restrict__ g,
                const float* __restrict__ bb, float* __restrict__ x) {
  __shared__ float red[8];
  const int tok = blockIdx.x, tid = threadIdx.x;
  const int s = tok & (Sq - 1);
  const int id = log_seqs[tok];
  float v = item_emb[(size_t)id * HID + tid] + pos_emb[(size_t)s * HID + tid];
  const float m = blockSum256(v, red) * (1.0f / HID);
  const float d = v - m;
  const float var = blockSum256(d * d, red) * (1.0f / HID);
  x[(size_t)tok * HID + tid] = d * rsqrtf(var + 1e-12f) * g[tid] + bb[tid];
}

// ------------- residual + layernorm: out = LN(x + y; g,b; eps) -------------
__global__ void __launch_bounds__(256)
resln_kernel(const float* __restrict__ x, const float* __restrict__ y,
             const float* __restrict__ g, const float* __restrict__ bb,
             float eps, float* __restrict__ out) {
  __shared__ float red[8];
  const int tok = blockIdx.x, tid = threadIdx.x;
  const size_t i = (size_t)tok * HID + tid;
  float v = x[i] + y[i];
  const float m = blockSum256(v, red) * (1.0f / HID);
  const float d = v - m;
  const float var = blockSum256(d * d, red) * (1.0f / HID);
  out[i] = d * rsqrtf(var + eps) * g[tid] + bb[tid];
}

// ---------------------------- x += a ---------------------------------------
__global__ void __launch_bounds__(256)
add_kernel(float* __restrict__ x, const float* __restrict__ a) {
  const size_t i = (size_t)blockIdx.x * HID + threadIdx.x;
  x[i] += a[i];
}

// -------- combine freq/spatial + std_norm (ddof=1, +1e-8 on std) ----------
__global__ void __launch_bounds__(256)
combine_stdnorm_kernel(const float* __restrict__ freq, const float* __restrict__ spat,
                       float* __restrict__ out) {
  __shared__ float red[8];
  const int tok = blockIdx.x, tid = threadIdx.x;
  const size_t i = (size_t)tok * HID + tid;
  float v = 0.9f * freq[i] + 0.1f * spat[i];
  const float m = blockSum256(v, red) * (1.0f / HID);
  const float d = v - m;
  const float var = blockSum256(d * d, red) * (1.0f / (HID - 1));
  out[i] = d / (sqrtf(var) + 1e-8f);
}

// ---------------------------------------------------------------------------
// WMMA GEMM:  C[M x N] = A[M x Kdim] * W[Kdim x N] + bias  (bf16 in, f32 acc)
// block = 128 threads (4 waves), tile BM=64 BN=64 BK=32.
// As[m][k] row-major; Bs[n][k] (weight tile TRANSPOSED).
// ---------------------------------------------------------------------------
__global__ void __launch_bounds__(128)
gemm_wmma(const float* __restrict__ A, const float* __restrict__ W,
          const float* __restrict__ bias, float* __restrict__ C,
          int Kdim, int N, int mode) {
  __shared__ alignas(16) __bf16 As[64][40];   // [m][k], pitch 80B
  __shared__ alignas(16) __bf16 Bs[64][40];   // [n][k], transposed weight tile
  const int tid = threadIdx.x;
  const int lane = tid & 31;
  const int wv = tid >> 5;        // 0..3
  const int half = lane >> 4;     // 0/1
  const int l16 = lane & 15;
  const int rowBase = blockIdx.y * 64;
  const int colBase = blockIdx.x * 64;
  f32x8 acc[4] = {};
  for (int k0 = 0; k0 < Kdim; k0 += 32) {
    __syncthreads();
    // A tile: 64x32 floats = 512 float4; 4 per thread; packed b64 LDS stores
    #pragma unroll
    for (int j = 0; j < 4; ++j) {
      const int idx4 = tid + j * 128;
      const int r = idx4 >> 3, c4 = (idx4 & 7) << 2;
      const float4 v = *(const float4*)&A[(size_t)(rowBase + r) * Kdim + k0 + c4];
      *(bf16x4*)&As[r][c4] = cvt4(v);
    }
    // B tile (transposed into Bs[n][k]): coalesced float4 reads along n
    #pragma unroll
    for (int j = 0; j < 4; ++j) {
      const int idx4 = tid + j * 128;
      const int r = idx4 >> 4, c4 = (idx4 & 15) << 2;   // r = k, c4 = n
      const float4 v = *(const float4*)&W[(size_t)(k0 + r) * N + colBase + c4];
      Bs[c4 + 0][r] = (__bf16)v.x;
      Bs[c4 + 1][r] = (__bf16)v.y;
      Bs[c4 + 2][r] = (__bf16)v.z;
      Bs[c4 + 3][r] = (__bf16)v.w;
    }
    __syncthreads();
    const int am = wv * 16 + l16;
    const bf16x16 af = ldfrag(&As[am][half * 8], &As[am][half * 8 + 16]);
    #pragma unroll
    for (int nt = 0; nt < 4; ++nt) {
      const int bn = nt * 16 + l16;
      const bf16x16 bf = ldfrag(&Bs[bn][half * 8], &Bs[bn][half * 8 + 16]);
      acc[nt] = __builtin_amdgcn_wmma_f32_16x16x32_bf16(
          false, af, false, bf, (short)0, acc[nt], false, false);
    }
  }
  #pragma unroll
  for (int nt = 0; nt < 4; ++nt) {
    #pragma unroll
    for (int r = 0; r < 8; ++r) {
      const int m = rowBase + wv * 16 + half * 8 + r;
      const int n = colBase + nt * 16 + l16;
      float v = acc[nt][r] + bias[n];
      if (mode == MODE_GELU)
        v = 0.5f * v * (1.0f + erff(v * 0.70710678118654752440f));
      if (mode == MODE_HEADS) {
        const int b = m >> 10, s = m & (Sq - 1);
        const int h = n >> 6,  d = n & (DHEAD - 1);
        C[(size_t)(((b * NHEAD + h) << 10) + s) * DHEAD + d] = v;
      } else {
        C[(size_t)m * N + n] = v;
      }
    }
  }
}

// ---------------------------------------------------------------------------
// Flash attention per (b,h): q,k,v are (B,NH,S,DH).  Block = 4 waves,
// 64 query rows per block, 32-key tiles, WMMA for S=QK^T and O+=P*V.
// Ks[t][d] natural layout; Vt[d][t] transposed on fill.
// mask = -10000 where !(t<=s && log_seqs[b,t]>0); scale 1/sqrt(64).
// ---------------------------------------------------------------------------
__global__ void __launch_bounds__(128)
attn_kernel(const float* __restrict__ q, const float* __restrict__ k,
            const float* __restrict__ v, const int* __restrict__ log_seqs,
            float* __restrict__ spat) {
  __shared__ alignas(16) __bf16 Qs[64][72];   // [row][d]
  __shared__ alignas(16) __bf16 Ks[32][72];   // [t][d]
  __shared__ alignas(16) __bf16 Vt[64][40];   // [d][t]
  __shared__ alignas(16) __bf16 Ps[64][40];   // [row][t]
  const int tid = threadIdx.x;
  const int lane = tid & 31;
  const int wv = tid >> 5;
  const int half = lane >> 4;
  const int l16 = lane & 15;
  const int b = blockIdx.z, h = blockIdx.y;
  const int q0 = blockIdx.x * 64;
  const size_t bh = ((size_t)(b * NHEAD + h)) << 10;
  const float* qb = q + bh * DHEAD;
  const float* kb = k + bh * DHEAD;
  const float* vb = v + bh * DHEAD;

  // Q tile: 64x64 floats = 1024 float4; 8 per thread
  #pragma unroll
  for (int j = 0; j < 8; ++j) {
    const int idx4 = tid + j * 128;
    const int r = idx4 >> 4, c4 = (idx4 & 15) << 2;
    const float4 vq = *(const float4*)&qb[(size_t)(q0 + r) * DHEAD + c4];
    *(bf16x4*)&Qs[r][c4] = cvt4(vq);
  }
  __syncthreads();

  const int am = wv * 16 + l16;
  bf16x16 aq[2];
  #pragma unroll
  for (int kk = 0; kk < 2; ++kk)
    aq[kk] = ldfrag(&Qs[am][kk * 32 + half * 8], &Qs[am][kk * 32 + half * 8 + 16]);

  float mrun[8], lrun[8];
  #pragma unroll
  for (int r = 0; r < 8; ++r) { mrun[r] = -3.0e38f; lrun[r] = 0.f; }
  f32x8 oacc[4] = {};

  for (int t0 = 0; t0 < Sq; t0 += 32) {
    __syncthreads();
    // K tile: 32x64 floats = 512 float4; 4 per thread; packed stores
    #pragma unroll
    for (int j = 0; j < 4; ++j) {
      const int idx4 = tid + j * 128;
      const int t = idx4 >> 4, c4 = (idx4 & 15) << 2;
      const float4 vk = *(const float4*)&kb[(size_t)(t0 + t) * DHEAD + c4];
      *(bf16x4*)&Ks[t][c4] = cvt4(vk);
    }
    // V tile transposed into Vt[d][t]: float4 reads, scalar scatter stores
    #pragma unroll
    for (int j = 0; j < 4; ++j) {
      const int idx4 = tid + j * 128;
      const int t = idx4 >> 4, c4 = (idx4 & 15) << 2;
      const float4 vv = *(const float4*)&vb[(size_t)(t0 + t) * DHEAD + c4];
      Vt[c4 + 0][t] = (__bf16)vv.x;
      Vt[c4 + 1][t] = (__bf16)vv.y;
      Vt[c4 + 2][t] = (__bf16)vv.z;
      Vt[c4 + 3][t] = (__bf16)vv.w;
    }
    __syncthreads();

    f32x8 sc[2] = {};
    #pragma unroll
    for (int kk = 0; kk < 2; ++kk) {
      #pragma unroll
      for (int nt = 0; nt < 2; ++nt) {
        const int tcol = nt * 16 + l16;
        const bf16x16 bk_ = ldfrag(&Ks[tcol][kk * 32 + half * 8],
                                   &Ks[tcol][kk * 32 + half * 8 + 16]);
        sc[nt] = __builtin_amdgcn_wmma_f32_16x16x32_bf16(
            false, aq[kk], false, bk_, (short)0, sc[nt], false, false);
      }
    }

    const int sqbase = q0 + wv * 16 + half * 8;
    float rm[8];
    #pragma unroll
    for (int r = 0; r < 8; ++r) {
      const int srow = sqbase + r;
      #pragma unroll
      for (int nt = 0; nt < 2; ++nt) {
        const int t = t0 + nt * 16 + l16;
        float val = sc[nt][r] * 0.125f;
        const bool ok = (t <= srow) && (log_seqs[(b << 10) + t] > 0);
        val += ok ? 0.f : -10000.f;
        sc[nt][r] = val;
      }
      float mx = fmaxf(sc[0][r], sc[1][r]);
      #pragma unroll
      for (int off = 8; off; off >>= 1) mx = fmaxf(mx, __shfl_xor(mx, off, 16));
      rm[r] = mx;
    }
    #pragma unroll
    for (int r = 0; r < 8; ++r) {
      const float nm = fmaxf(mrun[r], rm[r]);
      const float rescale = expf(mrun[r] - nm);
      const float p0 = expf(sc[0][r] - nm);
      const float p1 = expf(sc[1][r] - nm);
      float rs = p0 + p1;
      #pragma unroll
      for (int off = 8; off; off >>= 1) rs += __shfl_xor(rs, off, 16);
      lrun[r] = lrun[r] * rescale + rs;
      mrun[r] = nm;
      #pragma unroll
      for (int nt = 0; nt < 4; ++nt) oacc[nt][r] *= rescale;
      const int prow = wv * 16 + half * 8 + r;
      Ps[prow][l16]      = (__bf16)p0;
      Ps[prow][16 + l16] = (__bf16)p1;
    }
    __syncthreads();

    const bf16x16 ap = ldfrag(&Ps[am][half * 8], &Ps[am][half * 8 + 16]);
    #pragma unroll
    for (int nt = 0; nt < 4; ++nt) {
      const int dcol = nt * 16 + l16;
      const bf16x16 bv_ = ldfrag(&Vt[dcol][half * 8], &Vt[dcol][half * 8 + 16]);
      oacc[nt] = __builtin_amdgcn_wmma_f32_16x16x32_bf16(
          false, ap, false, bv_, (short)0, oacc[nt], false, false);
    }
  }

  #pragma unroll
  for (int r = 0; r < 8; ++r) {
    const float inv = 1.0f / lrun[r];
    const int srow = q0 + wv * 16 + half * 8 + r;
    #pragma unroll
    for (int nt = 0; nt < 4; ++nt) {
      const int d = nt * 16 + l16;
      spat[(size_t)((b << 10) + srow) * HID + h * DHEAD + d] = oacc[nt][r] * inv;
    }
  }
}

// ---------------------------------------------------------------------------
// Cross-spectrum:  spec[b][f] += (1/64) * sum_{s} rfft(q)[f] * conj(rfft(k)[f])
// ---------------------------------------------------------------------------
__global__ void __launch_bounds__(256)
corr_kernel(const float* __restrict__ q, const float* __restrict__ k,
            float* __restrict__ spec, int fstart, int fend) {
  __shared__ float c64[64], s64[64];
  __shared__ float red[16];
  const int tid = threadIdx.x;
  if (tid < 64) {
    const float ang = 6.28318530717958647692f * (float)tid / 64.0f;
    c64[tid] = cosf(ang); s64[tid] = sinf(ang);
  }
  __syncthreads();
  const int bh = blockIdx.x;
  const int b = bh >> 2;
  const float* qb = q + (((size_t)bh) << 10) * DHEAD;
  const float* kb = k + (((size_t)bh) << 10) * DHEAD;
  for (int f = fstart; f < fend; ++f) {
    float accRe = 0.f, accIm = 0.f;
    for (int s = tid; s < Sq; s += 256) {
      const float* qr_ = qb + (size_t)s * DHEAD;
      const float* kr_ = kb + (size_t)s * DHEAD;
      float qre = 0.f, qim = 0.f, kre = 0.f, kim = 0.f;
      int idx = 0;
      for (int d = 0; d < DHEAD; ++d) {
        const float c = c64[idx], sn = s64[idx];
        const float qv = qr_[d], kv = kr_[d];
        qre += qv * c; qim -= qv * sn;
        kre += kv * c; kim -= kv * sn;
        idx = (idx + f) & 63;
      }
      accRe += qre * kre + qim * kim;       // qf * conj(kf)
      accIm += qim * kre - qre * kim;
    }
    accRe *= (1.0f / 64.0f);                // ortho rfft: (1/8)^2
    accIm *= (1.0f / 64.0f);
    #pragma unroll
    for (int off = 16; off; off >>= 1) {
      accRe += __shfl_xor(accRe, off, 32);
      accIm += __shfl_xor(accIm, off, 32);
    }
    const int wv = tid >> 5, lane = tid & 31;
    if (lane == 0) { red[wv] = accRe; red[8 + wv] = accIm; }
    __syncthreads();
    if (tid == 0) {
      float r = 0.f, im = 0.f;
      for (int w = 0; w < 8; ++w) { r += red[w]; im += red[8 + w]; }
      atomicAdd(&spec[(b * NFREQ + f) * 2 + 0], r);
      atomicAdd(&spec[(b * NFREQ + f) * 2 + 1], im);
    }
    __syncthreads();
  }
}

// ---------------------------------------------------------------------------
// irfft(n=1024, ortho) of mean spectrum, top-69, softmax.  One block / batch.
// ---------------------------------------------------------------------------
__global__ void __launch_bounds__(256)
topk_kernel(const float* __restrict__ spec, float* __restrict__ wsel,
            int* __restrict__ isel) {
  __shared__ float corr[Sq];
  __shared__ float sval[256];
  __shared__ int   sidx[256];
  __shared__ float wtmp[TOPK];
  const int b = blockIdx.x, tid = threadIdx.x;
  const float* sp = spec + b * (NFREQ * 2);
  for (int t = tid; t < Sq; t += 256) {
    float acc = sp[0];
    for (int f = 1; f < NFREQ; ++f) {
      const float re = sp[2 * f], im = sp[2 * f + 1];
      const float ang = 6.28318530717958647692f * (float)(f * t) * (1.0f / 1024.0f);
      acc += 2.0f * (re * cosf(ang) - im * sinf(ang));
    }
    corr[t] = acc * (1.0f / 131072.0f);      // 1/(4096*32)
  }
  __syncthreads();
  for (int ks = 0; ks < TOPK; ++ks) {
    float best = -3.0e38f; int bi = 0;
    for (int t = tid; t < Sq; t += 256) {
      const float vv = corr[t];
      if (vv > best) { best = vv; bi = t; }
    }
    sval[tid] = best; sidx[tid] = bi;
    __syncthreads();
    for (int off = 128; off; off >>= 1) {
      if (tid < off) {
        if (sval[tid + off] > sval[tid]) { sval[tid] = sval[tid + off]; sidx[tid] = sidx[tid + off]; }
      }
      __syncthreads();
    }
    if (tid == 0) {
      wsel[b * TOPK + ks] = sval[0];
      isel[b * TOPK + ks] = sidx[0];
      corr[sidx[0]] = -3.0e38f;
    }
    __syncthreads();
  }
  if (tid == 0) {
    float mx = -3.0e38f;
    for (int i = 0; i < TOPK; ++i) mx = fmaxf(mx, wsel[b * TOPK + i]);
    float sm = 0.f;
    for (int i = 0; i < TOPK; ++i) { const float e = expf(wsel[b * TOPK + i] - mx); wtmp[i] = e; sm += e; }
    const float inv = 1.0f / sm;
    for (int i = 0; i < TOPK; ++i) wsel[b * TOPK + i] = wtmp[i] * inv;
  }
}

// ----- freq aggregation: freq[b,s,h*64+d] = sum_k w[b,k] * v[b,h,(s+idx)%S,d]
__global__ void __launch_bounds__(256)
agg_kernel(const float* __restrict__ v, const float* __restrict__ wsel,
           const int* __restrict__ isel, float* __restrict__ freq) {
  const int blk = blockIdx.x;            // 0..4095
  const int bh = blk >> 8;
  const int sc = blk & 255;
  const int tid = threadIdx.x;
  const int s = sc * 4 + (tid >> 6);
  const int d = tid & 63;
  const int b = bh >> 2, h = bh & 3;
  const float* vb = v + (((size_t)bh) << 10) * DHEAD;
  float acc = 0.f;
  for (int kk = 0; kk < TOPK; ++kk) {
    const float w = wsel[b * TOPK + kk];
    const int t = (s + isel[b * TOPK + kk]) & (Sq - 1);
    acc += w * vb[(size_t)t * DHEAD + d];
  }
  freq[(size_t)((b << 10) + s) * HID + h * DHEAD + d] = acc;
}

// -------- final logits: dot(final[b], item_emb[seq[b,j]]), wave per dot -----
__global__ void __launch_bounds__(256)
logits_kernel(const float* __restrict__ x, const float* __restrict__ item_emb,
              const int* __restrict__ pos, const int* __restrict__ neg,
              float* __restrict__ out) {
  const int gw = (blockIdx.x * blockDim.x + threadIdx.x) >> 5;
  const int lane = threadIdx.x & 31;
  if (gw >= 2 * Bq * Sq) return;
  const int sel = gw >> 12;
  const int rem = gw & 4095;
  const int b = rem >> 10, j = rem & 1023;
  const int id = (sel ? neg : pos)[b * Sq + j];
  const float* f = x + (size_t)(b * Sq + (Sq - 1)) * HID;
  const float* e = item_emb + (size_t)id * HID;
  float acc = 0.f;
  #pragma unroll
  for (int d = lane; d < HID; d += 32) acc += f[d] * e[d];
  #pragma unroll
  for (int off = 16; off; off >>= 1) acc += __shfl_xor(acc, off, 32);
  if (lane == 0) out[sel * 4096 + rem] = acc;
}

// ---------------------------------------------------------------------------
// Host orchestration.  Workspace layout (floats), total ~34 MB:
//   X 1M | Q 1M | K 1M | V 1M | SP 1M | FR 1M | CB 1M | T2 1M | spec/topk
//   T1 (FFN hidden, 4M) aliases Q..SP (dead by FFN time).
// ---------------------------------------------------------------------------
extern "C" void kernel_launch(void* const* d_in, const int* in_sizes, int n_in,
                              void* d_out, int out_size, void* d_ws, size_t ws_size,
                              hipStream_t stream) {
  (void)in_sizes; (void)n_in; (void)out_size; (void)ws_size;
  const int* log_seqs = (const int*)d_in[1];
  const int* pos_seqs = (const int*)d_in[2];
  const int* neg_seqs = (const int*)d_in[3];
  const float* item_emb = (const float*)d_in[4];
  const float* ln_b = (const float*)d_in[37];
  const float* ln_g = (const float*)d_in[38];
  const float* pos_emb = (const float*)d_in[39];

  float* ws = (float*)d_ws;
  const size_t M1 = (size_t)1 << 20;         // 1M floats
  float* X  = ws;
  float* Q  = ws + 1 * M1;
  float* Kb = ws + 2 * M1;
  float* V  = ws + 3 * M1;
  float* SP = ws + 4 * M1;
  float* T1 = Q;                              // 4M floats, aliases Q..SP
  float* FR = ws + 5 * M1;
  float* CB = ws + 6 * M1;
  float* T2 = ws + 7 * M1;
  float* SPEC = ws + 8 * M1;                  // B*33*2 = 264 floats
  float* WSEL = SPEC + 512;                   // B*69
  int*   ISEL = (int*)(SPEC + 1024);          // B*69

  const int TOK = Bq * Sq;                    // 4096 tokens
  embed_ln_kernel<<<TOK, 256, 0, stream>>>(log_seqs, item_emb, pos_emb, ln_g, ln_b, X);

  for (int l = 0; l < 2; ++l) {
    const float* const* P = (const float* const*)(d_in + 5 + 16 * l);
    // sorted-key order within each layer dict:
    const float* b1 = P[0],  *b2 = P[1],  *bk = P[2],  *bo = P[3];
    const float* bqv = P[4], *bv = P[5];
    const float* ln1_b = P[6], *ln1_g = P[7], *ln2_b = P[8], *ln2_g = P[9];
    const float* w1 = P[10], *w2 = P[11], *wk = P[12], *wo = P[13];
    const float* wq = P[14], *wv = P[15];

    // QKV projections (head-split epilogue)
    gemm_wmma<<<dim3(4, 64), 128, 0, stream>>>(X, wq, bqv, Q,  HID, HID, MODE_HEADS);
    gemm_wmma<<<dim3(4, 64), 128, 0, stream>>>(X, wk, bk,  Kb, HID, HID, MODE_HEADS);
    gemm_wmma<<<dim3(4, 64), 128, 0, stream>>>(X, wv, bv,  V,  HID, HID, MODE_HEADS);

    // spatial attention (flash, WMMA)
    attn_kernel<<<dim3(Sq / 64, NHEAD, Bq), 128, 0, stream>>>(Q, Kb, V, log_seqs, SP);

    // frequency branch: layer filter window (MAX_FREQ=513, win=487, slide=26)
    int fstart = 513 - 487 - l * 26; if (fstart < 0) fstart = 0;
    int fend = fstart + 487; if (fend > 513) fend = 513; if (fend > NFREQ) fend = NFREQ;
    hipMemsetAsync(SPEC, 0, (size_t)Bq * NFREQ * 2 * sizeof(float), stream);
    corr_kernel<<<Bq * NHEAD, 256, 0, stream>>>(Q, Kb, SPEC, fstart, fend);
    topk_kernel<<<Bq, 256, 0, stream>>>(SPEC, WSEL, ISEL);
    agg_kernel<<<TOK, 256, 0, stream>>>(V, WSEL, ISEL, FR);

    // combine + std_norm, output proj, residual LN, residual add
    combine_stdnorm_kernel<<<TOK, 256, 0, stream>>>(FR, SP, CB);
    gemm_wmma<<<dim3(4, 64), 128, 0, stream>>>(CB, wo, bo, T2, HID, HID, MODE_PLAIN);
    resln_kernel<<<TOK, 256, 0, stream>>>(X, T2, ln1_g, ln1_b, 1e-5f, CB);
    add_kernel<<<TOK, 256, 0, stream>>>(X, CB);

    // FFN: gelu(x@w1+b1)@w2+b2, LN, residual
    gemm_wmma<<<dim3(16, 64), 128, 0, stream>>>(X, w1, b1, T1, HID, 4 * HID, MODE_GELU);
    gemm_wmma<<<dim3(4, 64), 128, 0, stream>>>(T1, w2, b2, T2, 4 * HID, HID, MODE_PLAIN);
    resln_kernel<<<TOK, 256, 0, stream>>>(X, T2, ln2_g, ln2_b, 1e-5f, CB);
    add_kernel<<<TOK, 256, 0, stream>>>(X, CB);
  }

  logits_kernel<<<1024, 256, 0, stream>>>(X, item_emb, pos_seqs, neg_seqs, (float*)d_out);
}
